// SAGELANet_40991167873098
// MI455X (gfx1250) — compile-verified, matrix operand order checked by reading, not run on hardware
//
#include <hip/hip_runtime.h>
#include <hip/hip_bf16.h>
#include <hip/hip_fp16.h>

// SAGE-LA 2-layer GNN for MI455X (gfx1250, wave32).
// Roofline: edge scatter/gather dominates (~1.5 GB L2 traffic, ~123M f32
// atomics total); dense GEMMs are ~0.66 GFLOP -> use fp32 WMMA
// (V_WMMA_F32_16X16X4_F32) for exact fp32 math at zero marginal cost.

#define BATCH  16
#define NNODES 10000

typedef __attribute__((ext_vector_type(2)))  float    v2f;
typedef __attribute__((ext_vector_type(8)))  float    v8f;
typedef __attribute__((ext_vector_type(16))) _Float16 v16h;

// ---------------------------------------------------------------------------
// Kernel 1: per-node gate projections  gi[b,n] = x[b,n,:]·wg_i,  gj = x·wg_j
// Hoists the O(C) dot products out of the per-edge loop (E >> N).
// ---------------------------------------------------------------------------
__global__ __launch_bounds__(256) void gate_proj_kernel(
    const float* __restrict__ x,    // (B,N,C)
    const float* __restrict__ gw,   // (2C+1,) = [wg_i | wg_j | wg_e]
    float* __restrict__ gi,         // (B*N)
    float* __restrict__ gj,         // (B*N)
    int BN, int C)
{
  int i = blockIdx.x * blockDim.x + threadIdx.x;
  if (i >= BN) return;
  const float* xr = x + (size_t)i * C;
  float si = 0.f, sj = 0.f;
  for (int c = 0; c < C; ++c) {
    float v = xr[c];
    si += v * gw[c];
    sj += v * gw[C + c];
  }
  gi[i] = si;
  gj[i] = sj;
}

// ---------------------------------------------------------------------------
// Kernel 2: one wave per edge; lanes cover channels (x batches when C==16).
// msg = (e*amp_w[c]) * x_j * sigmoid(gi[dst]+gj[src]+e*wg_e+b)
// scatter-added into aggr with GLOBAL_ATOMIC_ADD_F32 (aggr resident in L2).
// ---------------------------------------------------------------------------
__global__ __launch_bounds__(256) void edge_scatter_kernel(
    const float* __restrict__ x,     // (B,N,C) source features
    const int*   __restrict__ eidx,  // (2,E): [src row | dst row]
    const float* __restrict__ ew,    // (E)
    const float* __restrict__ gi,    // (B*N) dst gate proj
    const float* __restrict__ gj,    // (B*N) src gate proj
    const float* __restrict__ ampw,  // (C)
    const float* __restrict__ gwe,   // &gate_w[2C] (scalar)
    const float* __restrict__ gb,    // gate bias (scalar)
    float* __restrict__ aggr,        // (B,N,C)
    float* __restrict__ cnt,         // (N)
    int E, int C)
{
  int wid  = (blockIdx.x * blockDim.x + threadIdx.x) >> 5;
  int lane = threadIdx.x & 31;
  if (wid >= E) return;

  int   src  = eidx[wid];
  int   dst  = eidx[E + wid];
  float e    = ew[wid];
  float elog = e * gwe[0] + gb[0];

  if (lane == 0) atomicAdd(&cnt[dst], 1.0f);

  int   c     = lane % C;       // lane -> channel
  int   bsub  = lane / C;       // packed batches per wave pass (C==16: 2)
  int   bstep = 32 / C;
  float ampc  = e * ampw[c];

  for (int b = bsub; b < BATCH; b += bstep) {
    float lamb = 1.f / (1.f + __expf(-(gi[b * NNODES + dst] +
                                       gj[b * NNODES + src] + elog)));
    float xj = x[((size_t)b * NNODES + src) * C + c];
    atomicAdd(&aggr[((size_t)b * NNODES + dst) * C + c], ampc * xj * lamb);
  }
}

// ---------------------------------------------------------------------------
// WMMA epilogue helper: C/D 16x16 f32 layout -> lane<16: M=r, N=lane;
// lane>=16: M=8+r, N=lane-16.   Bias + optional leaky-ReLU fused.
// ---------------------------------------------------------------------------
template <int COUT, int ACT>
__device__ __forceinline__ void store_tile(v8f acc, const float* __restrict__ bias,
                                           float* __restrict__ out,
                                           int b, int node0, int nt, int lane)
{
  int   ncol  = nt * 16 + (lane & 15);
  int   mbase = (lane >> 4) ? 8 : 0;
  float bv    = bias[ncol];
#pragma unroll
  for (int r = 0; r < 8; ++r) {
    float v = acc[r] + bv;
    if (ACT) v = (v > 0.f) ? v : 0.01f * v;
    out[((size_t)b * NNODES + node0 + mbase + r) * COUT + ncol] = v;
  }
}

// ---------------------------------------------------------------------------
// Kernel 3: normalize aggr, concat [x | aggr/max(cnt,1)], GEMM with WMMA.
// One wave per (batch, 16-node tile); A tile staged in LDS with +1 padding
// so the 16-lane column reads for the K=4 fragments are conflict-free.
// ---------------------------------------------------------------------------
template <int C, int COUT, int ACT>
__global__ __launch_bounds__(256) void norm_concat_gemm_wmma(
    const float* __restrict__ xdst,  // (B,N,C)
    const float* __restrict__ aggr,  // (B,N,C)
    const float* __restrict__ cnt,   // (N)
    const float* __restrict__ w,     // (2C, COUT) row-major
    const float* __restrict__ bias,  // (COUT)
    float* __restrict__ out)         // (B,N,COUT)
{
  constexpr int K  = 2 * C;
  constexpr int AS = K + 1;          // padded LDS row stride (bank-conflict free)
  __shared__ float sW[K * COUT];     // 4 KB both layers
  __shared__ float sA[8][16 * AS];   // per-wave A tile

  const int tid  = threadIdx.x;
  const int wid  = tid >> 5;
  const int lane = tid & 31;

  for (int i = tid; i < K * COUT; i += 256) sW[i] = w[i];
  __syncthreads();

  const int nTiles = NNODES / 16;    // 625, exact
  int g = blockIdx.x * 8 + wid;
  if (g >= BATCH * nTiles) return;
  int b     = g / nTiles;
  int node0 = (g % nTiles) * 16;

  // Build A tile: row m = [x_dst[node] (C) | aggr[node]/max(cnt,1) (C)]
  for (int idx = lane; idx < 16 * K; idx += 32) {
    int m = idx / K, k = idx % K;
    int node = node0 + m;
    float v;
    if (k < C) v = xdst[((size_t)b * NNODES + node) * C + k];
    else       v = aggr[((size_t)b * NNODES + node) * C + (k - C)] /
                   fmaxf(cnt[node], 1.f);
    sA[wid][m * AS + k] = v;
  }
  // Same-wave LDS write->read: DS ops in-order per wave; compiler waits dscnt.

  const int mrow = lane & 15;

#if __has_builtin(__builtin_amdgcn_wmma_f32_16x16x4_f32)
  // fp32 matrix path: exact fp32 GEMM via V_WMMA_F32_16X16X4_F32.
  // A 16x4 layout: lane<16 holds K={0,1} of row M=lane; lane>=16 K={2,3}.
  // B 4x16 layout: lane<16 holds N=lane of rows K={0,1}; lane>=16 K={2,3}.
  const int khalf = (lane >> 4) << 1;   // 0 or 2
#pragma unroll
  for (int nt = 0; nt < COUT / 16; ++nt) {
    v8f acc = {};
#pragma unroll
    for (int k0 = 0; k0 < K; k0 += 4) {
      v2f a, bm;
      a.x  = sA[wid][mrow * AS + k0 + khalf + 0];
      a.y  = sA[wid][mrow * AS + k0 + khalf + 1];
      bm.x = sW[(k0 + khalf + 0) * COUT + nt * 16 + mrow];
      bm.y = sW[(k0 + khalf + 1) * COUT + nt * 16 + mrow];
      acc  = __builtin_amdgcn_wmma_f32_16x16x4_f32(
                 false, a, false, bm, (short)0, acc, false, false);
    }
    store_tile<COUT, ACT>(acc, bias, out, b, node0, nt, lane);
  }
#else
  // Fallback: f16 inputs / f32 accum via confirmed wmma_f32_16x16x32_f16.
  // 16-bit A 16x32 layout: lane<16 -> halves0..7=K0..7, 8..15=K16..23;
  // lane>=16 -> K8..15 and K24..31 (K-half select = (lane>>4)*8).
  const int khi = (lane >> 4) * 8;
#pragma unroll
  for (int nt = 0; nt < COUT / 16; ++nt) {
    v8f acc = {};
#pragma unroll
    for (int k0 = 0; k0 < K; k0 += 32) {
      v16h af, bf;
#pragma unroll
      for (int i = 0; i < 8; ++i) {
        int kA = k0 + khi + i;
        int kB = k0 + 16 + khi + i;
        af[i]     = (_Float16)sA[wid][mrow * AS + kA];
        af[i + 8] = (_Float16)sA[wid][mrow * AS + kB];
        bf[i]     = (_Float16)sW[kA * COUT + nt * 16 + mrow];
        bf[i + 8] = (_Float16)sW[kB * COUT + nt * 16 + mrow];
      }
      acc = __builtin_amdgcn_wmma_f32_16x16x32_f16(
                false, af, false, bf, (short)0, acc, false, false);
    }
    store_tile<COUT, ACT>(acc, bias, out, b, node0, nt, lane);
  }
#endif
}

// ---------------------------------------------------------------------------
// Host launch: layer1 (C=32 -> 16, leaky-ReLU) then layer2 (16 -> 32).
// Layer-2 scratch aliases layer-1's gi/gj/cnt/aggr region (consumed by then).
// ---------------------------------------------------------------------------
extern "C" void kernel_launch(void* const* d_in, const int* in_sizes, int n_in,
                              void* d_out, int out_size, void* d_ws, size_t ws_size,
                              hipStream_t stream) {
  const float* X     = (const float*)d_in[0];
  const int*   ei1   = (const int*)  d_in[1];
  const float* ew1   = (const float*)d_in[2];
  // d_in[3] res_n_id1 == arange -> identity, unused
  const int*   ei2   = (const int*)  d_in[4];
  const float* ew2   = (const float*)d_in[5];
  // d_in[6] res_n_id2 unused
  const float* ampw1 = (const float*)d_in[7];
  const float* gw1   = (const float*)d_in[8];   // 65 = [wg_i(32)|wg_j(32)|wg_e(1)]
  const float* gb1   = (const float*)d_in[9];
  const float* w1    = (const float*)d_in[10];  // (64,16)
  const float* b1    = (const float*)d_in[11];
  const float* ampw2 = (const float*)d_in[12];
  const float* gw2   = (const float*)d_in[13];  // 33 = [wg_i(16)|wg_j(16)|wg_e(1)]
  const float* gb2   = (const float*)d_in[14];
  const float* w2    = (const float*)d_in[15];  // (32,32)
  const float* b2    = (const float*)d_in[16];
  float* out = (float*)d_out;

  const int E1 = in_sizes[2];
  const int E2 = in_sizes[5];
  const int BN = BATCH * NNODES;

  // Workspace layout (floats): gi | gj | cnt | aggr(BN*32) | h(BN*16)
  float* ws   = (float*)d_ws;
  float* gi   = ws;
  float* gj   = gi + BN;
  float* cnt  = gj + BN;
  float* aggr = cnt + NNODES;
  float* h    = aggr + (size_t)BN * 32;

  const int gpBlocks    = (BN + 255) / 256;
  const int wmmaBlocks  = (BATCH * (NNODES / 16) + 7) / 8;

  // ---------------- layer 1: C=32 -> C_SP=16, leaky ReLU ----------------
  hipMemsetAsync(cnt, 0, (NNODES + (size_t)BN * 32) * sizeof(float), stream);
  gate_proj_kernel<<<gpBlocks, 256, 0, stream>>>(X, gw1, gi, gj, BN, 32);
  edge_scatter_kernel<<<(E1 * 32 + 255) / 256, 256, 0, stream>>>(
      X, ei1, ew1, gi, gj, ampw1, gw1 + 64, gb1, aggr, cnt, E1, 32);
  norm_concat_gemm_wmma<32, 16, 1><<<wmmaBlocks, 256, 0, stream>>>(
      X, aggr, cnt, w1, b1, h);

  // ---------------- layer 2: C_SP=16 -> C_OUT=32 ----------------
  hipMemsetAsync(cnt, 0, (NNODES + (size_t)BN * 16) * sizeof(float), stream);
  gate_proj_kernel<<<gpBlocks, 256, 0, stream>>>(h, gw2, gi, gj, BN, 16);
  edge_scatter_kernel<<<(E2 * 32 + 255) / 256, 256, 0, stream>>>(
      h, ei2, ew2, gi, gj, ampw2, gw2 + 32, gb2, aggr, cnt, E2, 16);
  norm_concat_gemm_wmma<16, 32, 0><<<wmmaBlocks, 256, 0, stream>>>(
      h, aggr, cnt, w2, b2, out);
}
